// DistanceWeightedSampling_11501922418895
// MI455X (gfx1250) — matrix operand order repaired and checked by reading, not compiled, not run on hardware
//
#include <hip/hip_runtime.h>
#include <hip/hip_bf16.h>
#include <stdint.h>

// ---------------- problem constants ----------------
#define N_PTS   4096
#define DIM     128
#define KBLK    8
#define KM1     7                      // k-1 negatives per row
#define NPAIR   (N_PTS * KM1)          // 28672 triples
// output layout (flat f32, reference tuple order)
#define OFF0    0                       // a_indices            [28672]
#define OFF1    ((size_t)28672)         // x[a]                 [28672*128]
#define OFF2    ((size_t)3698688)       // x[p]
#define OFF3    ((size_t)7368704)       // x[n]
#define OFF4    ((size_t)11038720)      // x passthrough        [4096*128]
// workspace layout (bytes)
#define WS_XH    ((size_t)0)            // bf16 x: 4096*128*2 = 1048576 B
#define WS_SLOTS ((size_t)1048576)      // 2 x u32 (ordered dmin/dmax)
#define WS_BEST  ((size_t)1048584)      // u64[4096*7] running argmax

typedef __attribute__((ext_vector_type(16))) __bf16 v16bf;
typedef __attribute__((ext_vector_type(8)))  __bf16 v8bf;
typedef __attribute__((ext_vector_type(8)))  float  v8f;

struct BfPair { v8bf lo, hi; };

// monotone float <-> uint mapping so integer atomic min/max order floats
__device__ __forceinline__ unsigned int encOrd(float f) {
  unsigned int u = __float_as_uint(f);
  return (u & 0x80000000u) ? ~u : (u | 0x80000000u);
}
__device__ __forceinline__ float decOrd(unsigned int u) {
  unsigned int b = (u & 0x80000000u) ? (u & 0x7FFFFFFFu) : ~u;
  return __uint_as_float(b);
}

__device__ __forceinline__ float lwOf(float d) {
  // (2-128)*log(d) - 62.5*log(max(1 - d^2/4, 1e-8))
  return -126.0f * __logf(d) - 62.5f * __logf(fmaxf(1.0f - 0.25f * d * d, 1e-8f));
}

// Load the wave's A fragments (16 rows i0.., all K=128) once.
// A 16x32 bf16 layout: lanes0-15 hold K {0..7,16..23}, lanes16-31 {8..15,24..31}.
__device__ __forceinline__ void loadA(const __bf16* __restrict__ xh, int i0,
                                      int lane, v16bf a[4]) {
  const int lr   = lane & 15;
  const int hi16 = lane >> 4;
  const __bf16* arow = xh + (size_t)(i0 + lr) * DIM;
#pragma unroll
  for (int kc = 0; kc < 4; ++kc) {
    const int k0 = kc * 32;
    v8bf alo = *(const v8bf*)(arow + k0 + hi16 * 8);
    v8bf ahi = *(const v8bf*)(arow + k0 + 16 + hi16 * 8);
    BfPair ap{alo, ahi};
    a[kc] = __builtin_bit_cast(v16bf, ap);
  }
}

// 16x16 tile of sim = X X^T: issue all 8 B loads first, then 4 chained
// v_wmma_f32_16x16x32_bf16 so the compiler can overlap load latency.
__device__ __forceinline__ v8f simTile(const __bf16* __restrict__ xh,
                                       const v16bf a[4], int j0, int lane) {
  const int lr   = lane & 15;
  const int hi16 = lane >> 4;
  const __bf16* brow = xh + (size_t)(j0 + lr) * DIM + hi16 * 8;
  v16bf b0 = __builtin_bit_cast(v16bf, (BfPair{*(const v8bf*)(brow), *(const v8bf*)(brow + 16)}));        // K   0..31 slice for this lane
  v16bf b1 = __builtin_bit_cast(v16bf, (BfPair{*(const v8bf*)(brow + 32), *(const v8bf*)(brow + 48)}));   // K  32..63
  v16bf b2 = __builtin_bit_cast(v16bf, (BfPair{*(const v8bf*)(brow + 64), *(const v8bf*)(brow + 80)}));   // K  64..95
  v16bf b3 = __builtin_bit_cast(v16bf, (BfPair{*(const v8bf*)(brow + 96), *(const v8bf*)(brow + 112)}));   // K  96..127
  v8f acc = {};
  acc = __builtin_amdgcn_wmma_f32_16x16x32_bf16(false, a[0], false, b0, (short)0, acc, false, false);
  acc = __builtin_amdgcn_wmma_f32_16x16x32_bf16(false, a[1], false, b1, (short)0, acc, false, false);
  acc = __builtin_amdgcn_wmma_f32_16x16x32_bf16(false, a[2], false, b2, (short)0, acc, false, false);
  acc = __builtin_amdgcn_wmma_f32_16x16x32_bf16(false, a[3], false, b3, (short)0, acc, false, false);
  return acc;
}

// ---------------- K0: convert/copy/init ----------------
__global__ void __launch_bounds__(256)
k0_prep(const float* __restrict__ x, unsigned short* __restrict__ xh_raw,
        float* __restrict__ out, unsigned int* __restrict__ slots,
        unsigned long long* __restrict__ gBest) {
  __bf16* xh = (__bf16*)xh_raw;
  const int idx = blockIdx.x * 256 + threadIdx.x;      // 0 .. 524287
  const float v = x[idx];
  xh[idx] = (__bf16)v;
  out[OFF4 + (size_t)idx] = v;                          // x passthrough output
  if (idx < NPAIR) gBest[idx] = 0ull;                   // any real key > 0
  if (idx == 0) { slots[0] = 0xFFFFFFFFu; slots[1] = 0u; }  // dmin=+big, dmax=-big
}

// ---------------- K1: GEMM pass for distance extrema ----------------
// lw(d) is U-shaped in d, so max over the matrix = max(lw(dmin), lw(dmax)).
__global__ void __launch_bounds__(128)
k1_extrema(const unsigned short* __restrict__ xh_raw,
           unsigned int* __restrict__ slots) {
  const __bf16* xh = (const __bf16*)xh_raw;
  const int lane = threadIdx.x & 31;
  const int wave = threadIdx.x >> 5;
  const int i0 = blockIdx.x * 16;
  const int n  = lane & 15;
  const int mB = (lane >> 4) << 3;     // C layout: lanes16-31 hold M+8
  v16bf afrag[4];
  loadA(xh, i0, lane, afrag);
  float dmin = 3.0e38f, dmax = 0.0f;
  for (int t = 0; t < 8; ++t) {
    const int jt = (blockIdx.y * 4 + wave) * 8 + t;
    const int j0 = jt * 16;
    if (t + 1 < 8) {  // prefetch next tile's B row into WGP$/L2
      const __bf16* pre = xh + (size_t)(j0 + 16 + (lane & 15)) * DIM;
      __builtin_prefetch(pre, 0, 1);
    }
    v8f c = simTile(xh, afrag, j0, lane);
    if (j0 == i0) {    // uniform branch: eye only lives on the diagonal tile
#pragma unroll
      for (int v = 0; v < 8; ++v)
        if ((v + mB) == n) c[v] -= 0.5f;   // tq += 1  <=>  sim -= 0.5
    }
#pragma unroll
    for (int v = 0; v < 8; ++v) {
      float tq = __builtin_fmaf(-2.0f, c[v], 2.0f);
      float d  = fmaxf(__builtin_amdgcn_sqrtf(fmaxf(tq, 0.0f)), 0.5f);
      dmin = fminf(dmin, d);
      dmax = fmaxf(dmax, d);
    }
  }
#pragma unroll
  for (int off = 16; off >= 1; off >>= 1) {
    dmin = fminf(dmin, __shfl_xor(dmin, off, 32));
    dmax = fmaxf(dmax, __shfl_xor(dmax, off, 32));
  }
  if (lane == 0) {
    atomicMin(&slots[0], encOrd(dmin));
    atomicMax(&slots[1], encOrd(dmax));
  }
}

// ---------------- K2: GEMM + weights + Gumbel-race argmax sampling ----------------
// No eye needed here: the diagonal is inside the same-block mask -> w = 1e-8.
__global__ void __launch_bounds__(128)
k2_sample(const unsigned short* __restrict__ xh_raw,
          const unsigned int* __restrict__ slots,
          unsigned long long* __restrict__ gBest) {
  const __bf16* xh = (const __bf16*)xh_raw;
  __shared__ unsigned long long lBest[16 * KM1];
  for (int t = threadIdx.x; t < 16 * KM1; t += 128) lBest[t] = 0ull;
  __syncthreads();

  const float dmin = decOrd(slots[0]);
  const float dmax = decOrd(slots[1]);
  const float M = fmaxf(lwOf(dmin), lwOf(dmax));   // global max of log_weights

  const int lane = threadIdx.x & 31;
  const int wave = threadIdx.x >> 5;
  const int i0 = blockIdx.x * 16;
  const int n  = lane & 15;
  const int mB = (lane >> 4) << 3;
  v16bf afrag[4];
  loadA(xh, i0, lane, afrag);

  unsigned long long best[8][KM1];
#pragma unroll
  for (int v = 0; v < 8; ++v)
#pragma unroll
    for (int s = 0; s < KM1; ++s) best[v][s] = 0ull;

  for (int t = 0; t < 8; ++t) {
    const int jt = (blockIdx.y * 4 + wave) * 8 + t;
    const int j0 = jt * 16;
    const int j  = j0 + n;                        // column constant per lane/tile
    if (t + 1 < 8) {
      const __bf16* pre = xh + (size_t)(j0 + 16 + (lane & 15)) * DIM;
      __builtin_prefetch(pre, 0, 1);
    }
    v8f c = simTile(xh, afrag, j0, lane);
#pragma unroll
    for (int v = 0; v < 8; ++v) {
      const int i = i0 + v + mB;
      float tq = __builtin_fmaf(-2.0f, c[v], 2.0f);
      float d  = fmaxf(__builtin_amdgcn_sqrtf(fmaxf(tq, 0.0f)), 0.5f);
      const bool allowed = ((i >> 3) != (j >> 3)) && (d < 1.4f);
      const float w = allowed ? (__expf(lwOf(d) - M) + 1e-8f) : 1e-8f;
      const unsigned int h0 =
          (unsigned int)i * 0x9E3779B1u ^ (unsigned int)j * 0x85EBCA77u;
#pragma unroll
      for (int s = 0; s < KM1; ++s) {
        unsigned int h = h0 + (unsigned int)s * 0xC2B2AE3Du;
        h ^= h >> 16; h *= 0x7FEB352Du; h ^= h >> 15; h *= 0x846CA68Bu; h ^= h >> 16;
        const float u = ((float)(h >> 8) + 0.5f) * (1.0f / 16777216.0f); // (0,1)
        const float e = -__logf(u);                      // Exp(1)
        const float key = w * __builtin_amdgcn_rcpf(e);  // argmax key ~ sample ∝ w
        const unsigned long long cand =
            ((unsigned long long)encOrd(key) << 32) | (unsigned int)j;
        if (cand > best[v][s]) best[v][s] = cand;
      }
    }
  }
  // merge lanes -> LDS (ds_max_u64), then LDS -> global (global_atomic_max_u64)
#pragma unroll
  for (int v = 0; v < 8; ++v)
#pragma unroll
    for (int s = 0; s < KM1; ++s)
      atomicMax(&lBest[(v + mB) * KM1 + s], best[v][s]);
  __syncthreads();
  if (threadIdx.x < 16 * KM1) {
    const int r = threadIdx.x / KM1, s = threadIdx.x % KM1;
    atomicMax(&gBest[(size_t)(i0 + r) * KM1 + s], lBest[threadIdx.x]);
  }
}

// ---------------- K3: finalize indices + gather rows ----------------
__global__ void __launch_bounds__(128)
k3_gather(const float* __restrict__ x,
          const unsigned long long* __restrict__ gBest,
          float* __restrict__ out) {
  const int t = blockIdx.x;            // 0 .. 28671
  const int i = t / KM1;
  const int s = t - i * KM1;
  const int r = i & (KBLK - 1);
  const int p = (i & ~(KBLK - 1)) + s + ((s >= r) ? 1 : 0);
  const int nIdx = (int)(gBest[t] & 0xFFFFFFFFull);
  const int c = threadIdx.x;           // 0 .. 127 (one float each)
  out[OFF1 + (size_t)t * DIM + c] = x[(size_t)i * DIM + c];
  out[OFF2 + (size_t)t * DIM + c] = x[(size_t)p * DIM + c];
  out[OFF3 + (size_t)t * DIM + c] = x[(size_t)nIdx * DIM + c];
  if (c == 0) out[OFF0 + (size_t)t] = (float)i;  // a_indices as f32 output
}

extern "C" void kernel_launch(void* const* d_in, const int* in_sizes, int n_in,
                              void* d_out, int out_size, void* d_ws, size_t ws_size,
                              hipStream_t stream) {
  (void)in_sizes; (void)n_in; (void)out_size; (void)ws_size;
  const float* x = (const float*)d_in[0];
  float* out = (float*)d_out;
  char* ws = (char*)d_ws;
  unsigned short*      xh    = (unsigned short*)(ws + WS_XH);
  unsigned int*        slots = (unsigned int*)(ws + WS_SLOTS);
  unsigned long long*  gBest = (unsigned long long*)(ws + WS_BEST);

  k0_prep<<<dim3((N_PTS * DIM) / 256), dim3(256), 0, stream>>>(x, xh, out, slots, gBest);
  k1_extrema<<<dim3(N_PTS / 16, 8), dim3(128), 0, stream>>>(xh, slots);
  k2_sample<<<dim3(N_PTS / 16, 8), dim3(128), 0, stream>>>(xh, slots, gBest);
  k3_gather<<<dim3(NPAIR), dim3(128), 0, stream>>>(x, gBest, out);
}
